// Int8Linear_40905268527135
// MI455X (gfx1250) — compile-verified, matrix operand order checked by reading, not run on hardware
//
#include <hip/hip_runtime.h>

typedef __attribute__((ext_vector_type(4)))  float        v4f;
typedef __attribute__((ext_vector_type(8)))  float        v8f;
typedef __attribute__((ext_vector_type(4)))  __bf16       v4bf;
typedef __attribute__((ext_vector_type(8)))  __bf16       v8bf;
typedef __attribute__((ext_vector_type(16))) __bf16       v16bf;
typedef __attribute__((ext_vector_type(4)))  int          v4i;
typedef __attribute__((ext_vector_type(8)))  int          v8i;
typedef __attribute__((ext_vector_type(4)))  unsigned int v4u;

// Problem constants (match reference exactly)
constexpr int Kdim = 4096;
constexpr int Ndim = 11008;
constexpr int Mdim = 8192;           // BATCH * SEQ = 4 * 2048

#if __has_builtin(__builtin_amdgcn_tensor_load_to_lds)
#define HAS_TDM 1
#else
#define HAS_TDM 0
#endif

__device__ __forceinline__ __bf16 bf_rne(float f) {
  unsigned u = __builtin_bit_cast(unsigned, f);
  u += 0x7FFFu + ((u >> 16) & 1u);               // round-to-nearest-even
  return __builtin_bit_cast(__bf16, (unsigned short)(u >> 16));
}
__device__ __forceinline__ __bf16 bf_trunc_exact(float f) {
  // exact for integers |v| <= 255 (int8 weights)
  unsigned u = __builtin_bit_cast(unsigned, f);
  return __builtin_bit_cast(__bf16, (unsigned short)(u >> 16));
}

// ---------------------------------------------------------------------------
// Pre-pass 1: x f32 -> bf16 (RNE).  8 elems/thread.
// ---------------------------------------------------------------------------
__global__ __launch_bounds__(256)
void cvt_f32_to_bf16(const float* __restrict__ in, __bf16* __restrict__ out) {
  const long i = ((long)blockIdx.x * 256 + threadIdx.x) * 8;
  v4f a = *(const v4f*)(in + i);
  v4f b = *(const v4f*)(in + i + 4);
  v8bf t;
#pragma unroll
  for (int j = 0; j < 4; ++j) { t[j] = bf_rne(a[j]); t[4 + j] = bf_rne(b[j]); }
  *(v8bf*)(out + i) = t;
}

// ---------------------------------------------------------------------------
// Pre-pass 2: W int8 -> bf16 (exact).  16 elems/thread.
// ---------------------------------------------------------------------------
__global__ __launch_bounds__(256)
void cvt_i8_to_bf16(const signed char* __restrict__ in, __bf16* __restrict__ out) {
  const long i = ((long)blockIdx.x * 256 + threadIdx.x) * 16;
  v4i a = *(const v4i*)(in + i);
#pragma unroll
  for (int d = 0; d < 2; ++d) {
    v8bf t;
#pragma unroll
    for (int j = 0; j < 8; ++j) {
      const int b  = d * 8 + j;
      const int si = (signed char)(a[b >> 2] >> ((b & 3) * 8));
      t[j] = bf_trunc_exact((float)si);
    }
    *(v8bf*)(out + i + d * 8) = t;
  }
}

// ===========================================================================
// Main GEMM: bf16 operands from workspace, TDM tile staging, WMMA compute.
// Block tile 128(M) x 256(N) x 64(K); 8 waves, each computing a 64x64 tile
// (4x4 WMMA accumulators) -> 1.0 ds_load_b128 per v_wmma.
// ===========================================================================
constexpr int BM = 128, BN = 256, BK = 64;
constexpr int LDT = BK + 8;            // 72 bf16 = 144 B row stride; lane stride
                                       // 36 banks mod 64 -> 16 bank-groups 4 apart:
                                       // conflict-free ds_load_b128 fragments
constexpr int KT = Kdim / BK;          // 64 K-steps
constexpr int SMEM_BYTES = (2 * BM * LDT + 2 * BN * LDT) * 2;  // 110592 B

__device__ __forceinline__ unsigned lds_addr_of(const void* p) {
  return (unsigned)(unsigned long long)(__attribute__((address_space(3))) const void*)p;
}

#if HAS_TDM
// One TDM 2D-tile load: tile_dim0=64 bf16 (128 B rows), tile_dim1=tileRows,
// LDS padding 16 B after every 128 B -> LDS row stride 144 B (= LDT bf16).
__device__ __forceinline__ void tdm_load_tile(unsigned ldsByte, const __bf16* gtile,
                                              unsigned tensorRows, unsigned tileRows) {
  const unsigned long long ga = (unsigned long long)gtile;
  v4u g0;
  g0[0] = 1u;                                    // count=1, no gather
  g0[1] = ldsByte;                               // lds_addr
  g0[2] = (unsigned)(ga & 0xFFFFFFFFull);        // global_addr[31:0]
  g0[3] = (unsigned)((ga >> 32) & 0x01FFFFFFull) // global_addr[56:32]
          | (2u << 30);                          // type=2 (image)
  v8i g1;
  g1[0] = (int)((1u << 16)      // data_size = 2 B
              | (1u << 20)      // pad_enable
              | (4u << 22)      // pad_interval: 32 DW = 128 B
              | (3u << 25));    // pad_amount:    4 DW = 16 B
  g1[1] = (int)((Kdim & 0xFFFFu) << 16);                          // tensor_dim0 lo16
  g1[2] = (int)(((unsigned)Kdim >> 16) | ((tensorRows & 0xFFFFu) << 16));
  g1[3] = (int)((tensorRows >> 16) | ((unsigned)BK << 16));       // tile_dim0 = 64
  g1[4] = (int)tileRows;                                          // tile_dim1
  g1[5] = (int)Kdim;                                              // dim0_stride lo32
  g1[6] = 0;
  g1[7] = 0;
  const v4i z4 = {0, 0, 0, 0};
#if __clang_major__ >= 23
  const v8i z8 = {0, 0, 0, 0, 0, 0, 0, 0};
  __builtin_amdgcn_tensor_load_to_lds(g0, g1, z4, z4, z8, 0);
#else
  __builtin_amdgcn_tensor_load_to_lds(g0, g1, z4, z4, 0);
#endif
}
#endif

__global__ __launch_bounds__(256)
void gemm_bf16_wmma(const __bf16* __restrict__ xb,   // [8192][4096]
                    const __bf16* __restrict__ wb,   // [11008][4096]
                    const float* __restrict__ scale,
                    const float* __restrict__ bias,
                    float* __restrict__ out)
{
  extern __shared__ __bf16 smem[];                 // [2][BM*LDT] A, [2][BN*LDT] B

  const int tid   = threadIdx.x;
  const int lane  = tid & 31;
  const int wave  = tid >> 5;
  const int waveM = wave >> 2;                     // 0..1  -> M offset *64
  const int waveN = wave & 3;                      // 0..3  -> N offset *64

  const int n0 = blockIdx.x * BN;                  // 43 tiles
  const int m0 = blockIdx.y * BM;                  // 64 tiles

  __bf16* const sA = smem;                         // + buf*BM*LDT
  __bf16* const sB = smem + 2 * BM * LDT;          // + buf*BN*LDT

  auto stage = [&](int buf, int kt) {
#if HAS_TDM
    if (wave == 0) {
      tdm_load_tile(lds_addr_of(sA + buf * BM * LDT),
                    xb + (long)m0 * Kdim + kt * BK, (unsigned)Mdim, (unsigned)BM);
      tdm_load_tile(lds_addr_of(sB + buf * BN * LDT),
                    wb + (long)n0 * Kdim + kt * BK, (unsigned)Ndim, (unsigned)BN);
    }
#else
    const __bf16* gA = xb + (long)m0 * Kdim + kt * BK;
    const __bf16* gB = wb + (long)n0 * Kdim + kt * BK;
    __bf16* dA = sA + buf * BM * LDT;
    __bf16* dB = sB + buf * BN * LDT;
    for (int i = tid; i < BM * 8; i += 256) {      // 8 chunks of 8 bf16 per row
      const int r = i >> 3, c = (i & 7) * 8;
      *(v8bf*)&dA[r * LDT + c] = *(const v8bf*)(gA + (long)r * Kdim + c);
    }
    for (int i = tid; i < BN * 8; i += 256) {
      const int r = i >> 3, c = (i & 7) * 8;
      *(v8bf*)&dB[r * LDT + c] = *(const v8bf*)(gB + (long)r * Kdim + c);
    }
#endif
  };
  auto wait_stage = [&]() {
#if HAS_TDM
    if (wave == 0) {
#if __has_builtin(__builtin_amdgcn_s_wait_tensorcnt)
      __builtin_amdgcn_s_wait_tensorcnt(0);
#else
      asm volatile("s_wait_tensorcnt 0x0" ::: "memory");
#endif
    }
#endif
  };

  // Fragment offsets (16-bit A 16x32 layout): lane<16 -> K {0..7,16..23},
  // lane>=16 -> K {8..15,24..31}; per k-substep add ks*32.
  const int kHalf = (lane >> 4) * 8;
  int aoff[4], boff[4];
#pragma unroll
  for (int mi = 0; mi < 4; ++mi)
    aoff[mi] = (waveM * 64 + mi * 16 + (lane & 15)) * LDT + kHalf;
#pragma unroll
  for (int ni = 0; ni < 4; ++ni)
    boff[ni] = (waveN * 64 + ni * 16 + (lane & 15)) * LDT + kHalf;

  v8f acc[4][4] = {};

  stage(0, 0);
  wait_stage();
  __syncthreads();

  for (int kt = 0; kt < KT; ++kt) {
    const int cur = kt & 1;
    if (kt + 1 < KT) stage(cur ^ 1, kt + 1);       // async stage next tile

    const __bf16* A = sA + cur * BM * LDT;
    const __bf16* B = sB + cur * BN * LDT;

#pragma unroll
    for (int ks = 0; ks < 2; ++ks) {               // two 16x16x32 sub-steps
      const int kc = ks * 32;
      v16bf af[4], bfv[4];
#pragma unroll
      for (int mi = 0; mi < 4; ++mi) {
        v8bf lo = *(const v8bf*)(A + aoff[mi] + kc);
        v8bf hi = *(const v8bf*)(A + aoff[mi] + kc + 16);
        af[mi] = __builtin_shufflevector(lo, hi, 0,1,2,3,4,5,6,7,8,9,10,11,12,13,14,15);
      }
#pragma unroll
      for (int ni = 0; ni < 4; ++ni) {
        v8bf lo = *(const v8bf*)(B + boff[ni] + kc);
        v8bf hi = *(const v8bf*)(B + boff[ni] + kc + 16);
        bfv[ni] = __builtin_shufflevector(lo, hi, 0,1,2,3,4,5,6,7,8,9,10,11,12,13,14,15);
      }
#pragma unroll
      for (int mi = 0; mi < 4; ++mi)
#pragma unroll
        for (int ni = 0; ni < 4; ++ni)
          acc[mi][ni] = __builtin_amdgcn_wmma_f32_16x16x32_bf16(
              false, af[mi], false, bfv[ni], (short)0, acc[mi][ni], false, false);
    }

    if (kt + 1 < KT) wait_stage();
    __syncthreads();
  }

  // Epilogue: dequant scale + bias in f32.
  // C/D layout: lanes 0-15 -> N=lane, M=vgpr; lanes 16-31 -> N=lane-16, M=vgpr+8
  const int nlane = lane & 15;
  const int rsel  = (lane >> 4) * 8;
#pragma unroll
  for (int ni = 0; ni < 4; ++ni) {
    const int   n  = n0 + waveN * 64 + ni * 16 + nlane;
    const float s  = scale[n];
    const float bz = bias[n];
#pragma unroll
    for (int mi = 0; mi < 4; ++mi) {
      const long mbase = (long)(m0 + waveM * 64 + mi * 16 + rsel);
#pragma unroll
      for (int j = 0; j < 8; ++j)
        out[(mbase + j) * Ndim + n] = fmaf(acc[mi][ni][j], s, bz);
    }
  }
}

// ===========================================================================
// Fallback: fused single-kernel version (round-1) if ws too small
// ===========================================================================
constexpr int FBM = 128, FBN = 128, FBK = 32, FLDT = 40, FKT = Kdim / FBK;

__global__ __launch_bounds__(256)
void int8linear_wmma_fused(const float* __restrict__ x,
                           const signed char* __restrict__ w,
                           const float* __restrict__ scale,
                           const float* __restrict__ bias,
                           float* __restrict__ out)
{
  __shared__ __bf16 sA[2][FBM * FLDT];
  __shared__ __bf16 sB[2][FBN * FLDT];

  const int tid   = threadIdx.x;
  const int lane  = tid & 31;
  const int wave  = tid >> 5;
  const int waveM = wave >> 2;
  const int waveN = wave & 3;
  const int n0 = blockIdx.x * FBN;
  const int m0 = blockIdx.y * FBM;

  const int arow = tid >> 3;
  const int acol = (tid & 7) * 4;
  const int brow = tid >> 1;
  const int bcol = (tid & 1) * 16;

  const float*       xg = x + (long)m0 * Kdim;
  const signed char* wg = w + (long)n0 * Kdim;

  v4f aReg[4];
  v4i bReg;

  auto loadG = [&](int kt) {
    const int k0 = kt * FBK;
#pragma unroll
    for (int i = 0; i < 4; ++i)
      aReg[i] = *(const v4f*)(xg + (long)(arow + i * 32) * Kdim + k0 + acol);
    bReg = *(const v4i*)(wg + (long)brow * Kdim + k0 + bcol);
  };
  auto stageLDS = [&](int buf) {
#pragma unroll
    for (int i = 0; i < 4; ++i) {
      v4bf t;
      t[0] = bf_rne(aReg[i][0]); t[1] = bf_rne(aReg[i][1]);
      t[2] = bf_rne(aReg[i][2]); t[3] = bf_rne(aReg[i][3]);
      *(v4bf*)&sA[buf][(arow + i * 32) * FLDT + acol] = t;
    }
#pragma unroll
    for (int d = 0; d < 2; ++d) {
      v8bf t;
#pragma unroll
      for (int j = 0; j < 8; ++j) {
        const int b  = d * 8 + j;
        const int si = (signed char)(bReg[b >> 2] >> ((b & 3) * 8));
        t[j] = bf_trunc_exact((float)si);
      }
      *(v8bf*)&sB[buf][brow * FLDT + bcol + d * 8] = t;
    }
  };

  const int kHalf = (lane >> 4) * 8;
  int aoff[4], boff[2];
#pragma unroll
  for (int mi = 0; mi < 4; ++mi)
    aoff[mi] = (waveM * 64 + mi * 16 + (lane & 15)) * FLDT + kHalf;
#pragma unroll
  for (int ni = 0; ni < 2; ++ni)
    boff[ni] = (waveN * 32 + ni * 16 + (lane & 15)) * FLDT + kHalf;

  v8f acc[4][2] = {};
  loadG(0);
  stageLDS(0);
  __syncthreads();

  for (int kt = 0; kt < FKT; ++kt) {
    const int cur = kt & 1;
    if (kt + 1 < FKT) loadG(kt + 1);
    const __bf16* A = sA[cur];
    const __bf16* B = sB[cur];
    v16bf af[4], bfv[2];
#pragma unroll
    for (int mi = 0; mi < 4; ++mi) {
      v8bf lo = *(const v8bf*)(A + aoff[mi]);
      v8bf hi = *(const v8bf*)(A + aoff[mi] + 16);
      af[mi] = __builtin_shufflevector(lo, hi, 0,1,2,3,4,5,6,7,8,9,10,11,12,13,14,15);
    }
#pragma unroll
    for (int ni = 0; ni < 2; ++ni) {
      v8bf lo = *(const v8bf*)(B + boff[ni]);
      v8bf hi = *(const v8bf*)(B + boff[ni] + 16);
      bfv[ni] = __builtin_shufflevector(lo, hi, 0,1,2,3,4,5,6,7,8,9,10,11,12,13,14,15);
    }
#pragma unroll
    for (int mi = 0; mi < 4; ++mi)
#pragma unroll
      for (int ni = 0; ni < 2; ++ni)
        acc[mi][ni] = __builtin_amdgcn_wmma_f32_16x16x32_bf16(
            false, af[mi], false, bfv[ni], (short)0, acc[mi][ni], false, false);
    if (kt + 1 < FKT) stageLDS((kt + 1) & 1);
    __syncthreads();
  }

  const int nlane = lane & 15;
  const int rsel  = (lane >> 4) * 8;
#pragma unroll
  for (int ni = 0; ni < 2; ++ni) {
    const int   n  = n0 + waveN * 32 + ni * 16 + nlane;
    const float s  = scale[n];
    const float bz = bias[n];
#pragma unroll
    for (int mi = 0; mi < 4; ++mi) {
      const long mbase = (long)(m0 + waveM * 64 + mi * 16 + rsel);
#pragma unroll
      for (int j = 0; j < 8; ++j)
        out[(mbase + j) * Ndim + n] = fmaf(acc[mi][ni][j], s, bz);
    }
  }
}

// ===========================================================================
extern "C" void kernel_launch(void* const* d_in, const int* in_sizes, int n_in,
                              void* d_out, int out_size, void* d_ws, size_t ws_size,
                              hipStream_t stream) {
  const float*       x     = (const float*)d_in[0];
  const signed char* w     = (const signed char*)d_in[1];
  const float*       scale = (const float*)d_in[2];
  const float*       bias  = (const float*)d_in[3];
  float*             out   = (float*)d_out;

  const size_t needX = (size_t)Mdim * Kdim * sizeof(__bf16);   // 64 MiB
  const size_t needW = (size_t)Ndim * Kdim * sizeof(__bf16);   // 86 MiB

  if (ws_size >= needX + needW) {
    __bf16* xb = (__bf16*)d_ws;
    __bf16* wb = (__bf16*)((char*)d_ws + needX);
    cvt_f32_to_bf16<<<(Mdim * (long)Kdim) / (256 * 8), 256, 0, stream>>>(x, xb);
    cvt_i8_to_bf16<<<(Ndim * (long)Kdim) / (256 * 16), 256, 0, stream>>>(w, wb);
    dim3 grid(Ndim / BN, Mdim / BM);   // 43 x 64
    gemm_bf16_wmma<<<grid, dim3(256), SMEM_BYTES, stream>>>(xb, wb, scale, bias, out);
  } else {
    dim3 fgrid(Ndim / FBN, Mdim / FBM);  // 86 x 64
    int8linear_wmma_fused<<<fgrid, dim3(256), 0, stream>>>(x, w, scale, bias, out);
  }
}
